// QueryConditionedModel_3204045603594
// MI455X (gfx1250) — matrix operand order repaired and checked by reading, not compiled, not run on hardware
//
#include <hip/hip_runtime.h>
#include <hip/hip_bf16.h>
#include <stdint.h>

// ---------------------------------------------------------------------------
// QueryConditionedModel for MI455X (gfx1250, wave32, WMMA).
//
// Roofline: encoder FFN = 2*(64*128 + 128*64) flops * 1,048,576 tokens
//           ~= 34.4 GFLOP; HBM traffic ~8 MB  ->  compute (matrix-pipe) bound.
// Path:     f16 A/B with f32 accumulate via v_wmma_f32_16x16x32_f16.
//           Gate top-k only needs h . gate_w[:64] (monotone sigmoid + per-batch
//           constant), so [B,L,64] h is never materialized (saves 256 MB).
// ---------------------------------------------------------------------------

#define D     64
#define H     128
#define VOC   64
#define NB    256
#define LSEQ  4096
#define TOKW  16          // tokens per wave (WMMA M)
#define WAVES1 4          // waves per encoder block
#define TOKB  (WAVES1*TOKW)

typedef __attribute__((ext_vector_type(16))) _Float16 v16h;
typedef __attribute__((ext_vector_type(8)))  float    v8f;

union V16U { v16h h; uint32_t u[8]; };

__device__ __forceinline__ float wave_xor_add(float v, int m) {
  return v + __shfl_xor(v, m, 32);
}

// ------------------------------- zero scratch ------------------------------
__global__ void k_zero(float* __restrict__ p, int n) {
  int i = blockIdx.x * blockDim.x + threadIdx.x;
  if (i < n) p[i] = 0.0f;
}

// ------------------------------- encoder -----------------------------------
// grid = (B*L)/TOKB blocks, 128 threads (4 waves x 16 tokens).
__global__ __launch_bounds__(128) void k_encoder(
    const int*   __restrict__ seq,
    const float* __restrict__ embed,
    const float* __restrict__ w1, const float* __restrict__ b1,
    const float* __restrict__ w2, const float* __restrict__ b2,
    const float* __restrict__ lng, const float* __restrict__ lnb,
    const float* __restrict__ gw,          // gate_w[0:64]
    float* __restrict__ sgate,             // [B*L] gate scores
    float* __restrict__ gHsum)             // [B*64] sum_h accumulator (zeroed)
{
  __shared__ __align__(16) _Float16 sEmb[VOC * D];     // 8 KB
  __shared__ __align__(16) _Float16 sW1t[H * D];       // [n][k], 16 KB
  __shared__ __align__(16) _Float16 sW2t[D * H];       // [n][k], 16 KB
  __shared__ __align__(16) _Float16 sT16[WAVES1 * TOKW * H]; // 16 KB
  __shared__ float sB1[H], sB2[D], sLg[D], sLb[D], sGw[D], sHsum[D];

  const int tid = threadIdx.x;

  // Stage f32 -> f16 weights (transposed so B-fragment k-pairs are contiguous)
  for (int e = tid; e < VOC * D; e += 128) sEmb[e] = (_Float16)embed[e];
  for (int e = tid; e < H * D; e += 128) {
    int n = e / D, k = e - n * D;                      // sW1t[n][k] = W1[k][n]
    sW1t[e] = (_Float16)w1[k * H + n];
  }
  for (int e = tid; e < D * H; e += 128) {
    int n = e / H, k = e - n * H;                      // sW2t[n][k] = W2[k][n]
    sW2t[e] = (_Float16)w2[k * D + n];
  }
  if (tid < H) sB1[tid] = b1[tid];
  if (tid < D) {
    sB2[tid] = b2[tid]; sLg[tid] = lng[tid]; sLb[tid] = lnb[tid];
    sGw[tid] = gw[tid]; sHsum[tid] = 0.0f;
  }
  __syncthreads();

  const int wave = tid >> 5, lane = tid & 31;
  const int nl = lane & 15, half = lane >> 4;
  const int kb  = half * 8;     // A-fragment K base (16-bit A layout)
  const int kb2 = half * 16;    // unused; B fragments use kb, same k order as A
  const long gbase = (long)blockIdx.x * TOKB + wave * TOKW; // first token of wave

  // ---- A fragments for FFN1: rows = 16 tokens, K = 64 (2 steps of 32) ----
  const int id_a = seq[gbase + nl];
  V16U a1[2];
#pragma unroll
  for (int ks = 0; ks < 2; ++ks)
#pragma unroll
    for (int v = 0; v < 8; ++v) {
      int k = 32 * ks + ((v < 4) ? (kb + 2 * v) : (8 + kb + 2 * v));
      a1[ks].u[v] = *(const uint32_t*)&sEmb[id_a * D + k];
    }

  // ---- FFN1: t[16x128] = relu(h0 @ W1 + b1), 8 N-tiles x 2 K-steps -------
#pragma unroll
  for (int j = 0; j < 8; ++j) {
    v8f acc = {};
#pragma unroll
    for (int ks = 0; ks < 2; ++ks) {
      V16U bb;
#pragma unroll
      for (int v = 0; v < 8; ++v) {
        int k = 32 * ks + ((v < 4) ? (kb + 2 * v) : (8 + kb + 2 * v));
        bb.u[v] = *(const uint32_t*)&sW1t[(16 * j + nl) * D + k];
      }
      acc = __builtin_amdgcn_wmma_f32_16x16x32_f16(
          false, a1[ks].h, false, bb.h, (short)0, acc, false, false);
    }
    const int n = 16 * j + nl;
    const float bias = sB1[n];
#pragma unroll
    for (int r = 0; r < 8; ++r) {
      int m = r + 8 * half;                 // C/D layout row
      float tv = acc[r] + bias;
      tv = tv > 0.0f ? tv : 0.0f;
      sT16[(wave * TOKW + m) * H + n] = (_Float16)tv;
    }
  }
  __syncthreads();

  // ---- FFN2: y[16x64] = t @ W2 + b2, 4 N-tiles x 4 K-steps ---------------
  V16U a2[4];
#pragma unroll
  for (int ks = 0; ks < 4; ++ks)
#pragma unroll
    for (int v = 0; v < 8; ++v) {
      int k = 32 * ks + ((v < 4) ? (kb + 2 * v) : (8 + kb + 2 * v));
      a2[ks].u[v] = *(const uint32_t*)&sT16[(wave * TOKW + nl) * H + k];
    }

  float yv[4][8];
#pragma unroll
  for (int j = 0; j < 4; ++j) {
    v8f acc = {};
#pragma unroll
    for (int ks = 0; ks < 4; ++ks) {
      V16U bb;
#pragma unroll
      for (int v = 0; v < 8; ++v) {
        int k = 32 * ks + ((v < 4) ? (kb + 2 * v) : (8 + kb + 2 * v));
        bb.u[v] = *(const uint32_t*)&sW2t[(16 * j + nl) * H + k];
      }
      acc = __builtin_amdgcn_wmma_f32_16x16x32_f16(
          false, a2[ks].h, false, bb.h, (short)0, acc, false, false);
    }
    const float bias = sB2[16 * j + nl];
#pragma unroll
    for (int r = 0; r < 8; ++r) yv[j][r] = acc[r] + bias;
  }

  // ---- residual h0 (f32) + LayerNorm entirely in C-fragment registers ----
  int ids8[8];
#pragma unroll
  for (int r = 0; r < 8; ++r) ids8[r] = seq[gbase + r + 8 * half];
#pragma unroll
  for (int j = 0; j < 4; ++j) {
    const int n = 16 * j + nl;
#pragma unroll
    for (int r = 0; r < 8; ++r) yv[j][r] += embed[ids8[r] * D + n];
  }

  float s1[8], s2[8];
#pragma unroll
  for (int r = 0; r < 8; ++r) {
    s1[r] = 0.0f; s2[r] = 0.0f;
#pragma unroll
    for (int j = 0; j < 4; ++j) { s1[r] += yv[j][r]; s2[r] += yv[j][r] * yv[j][r]; }
  }
#pragma unroll
  for (int m = 1; m <= 8; m <<= 1)        // reduce across the 16-lane half
#pragma unroll
    for (int r = 0; r < 8; ++r) { s1[r] = wave_xor_add(s1[r], m); s2[r] = wave_xor_add(s2[r], m); }

  float mean[8], rstd[8];
#pragma unroll
  for (int r = 0; r < 8; ++r) {
    mean[r] = s1[r] * (1.0f / D);
    float var = s2[r] * (1.0f / D) - mean[r] * mean[r];
    rstd[r] = rsqrtf(var + 1e-5f);
  }

  // normalize in place; accumulate gate dot + column sums
  float gp[8];
#pragma unroll
  for (int r = 0; r < 8; ++r) gp[r] = 0.0f;
  float cs[4];
#pragma unroll
  for (int j = 0; j < 4; ++j) {
    const int n = 16 * j + nl;
    const float g = sLg[n], bconst = sLb[n], gwn = sGw[n];
    cs[j] = 0.0f;
#pragma unroll
    for (int r = 0; r < 8; ++r) {
      float hv = (yv[j][r] - mean[r]) * rstd[r] * g + bconst;
      yv[j][r] = hv;
      gp[r] += hv * gwn;
      cs[j] += hv;
    }
  }
#pragma unroll
  for (int m = 1; m <= 8; m <<= 1)
#pragma unroll
    for (int r = 0; r < 8; ++r) gp[r] = wave_xor_add(gp[r], m);
  if (nl == 0) {
#pragma unroll
    for (int r = 0; r < 8; ++r) sgate[gbase + r + 8 * half] = gp[r];
  }
#pragma unroll
  for (int j = 0; j < 4; ++j) {
    cs[j] = wave_xor_add(cs[j], 16);      // combine the two row-halves
    if (half == 0) atomicAdd(&sHsum[16 * j + nl], cs[j]);
  }

  __syncthreads();
  if (tid < D) {
    const int b = (int)(((long)blockIdx.x * TOKB) / LSEQ);
    atomicAdd(&gHsum[b * D + tid], sHsum[tid]);
  }
}

// ------------------------------- top-8 -------------------------------------
__global__ __launch_bounds__(256) void k_topk(const float* __restrict__ sgate,
                                              int* __restrict__ idxOut) {
  __shared__ float sv[LSEQ];
  __shared__ float rv[256];
  __shared__ int   ri[256];
  const int b = blockIdx.x, tid = threadIdx.x;
  for (int i = tid; i < LSEQ; i += 256) sv[i] = sgate[(long)b * LSEQ + i];
  __syncthreads();
  for (int round = 0; round < 8; ++round) {
    float best = -__builtin_inff(); int bi = 0;
    for (int i = tid; i < LSEQ; i += 256) {
      float v = sv[i];
      if (v > best) { best = v; bi = i; }
    }
    rv[tid] = best; ri[tid] = bi;
    __syncthreads();
    for (int s = 128; s > 0; s >>= 1) {
      if (tid < s) {
        float ov = rv[tid + s]; int oi = ri[tid + s];
        if (ov > rv[tid] || (ov == rv[tid] && oi < ri[tid])) { rv[tid] = ov; ri[tid] = oi; }
      }
      __syncthreads();
    }
    if (tid == 0) { idxOut[b * 8 + round] = ri[0]; sv[ri[0]] = -__builtin_inff(); }
    __syncthreads();
  }
}

// ------------------------------- readout -----------------------------------
// one block per batch; waves 0..7 recompute h for the 8 selected tokens,
// wave 8 for the last token; wave 0 then does type/softmax/attention/logits.
__global__ __launch_bounds__(288) void k_readout(
    const int* __restrict__ seq, const float* __restrict__ embed,
    const float* __restrict__ w1, const float* __restrict__ b1,
    const float* __restrict__ w2, const float* __restrict__ b2,
    const float* __restrict__ lng, const float* __restrict__ lnb,
    const float* __restrict__ tpw, const float* __restrict__ tpb,
    const float* __restrict__ qw,  const float* __restrict__ qb,
    const float* __restrict__ ow,  const float* __restrict__ ob,
    const float* __restrict__ gHsum, const int* __restrict__ idxArr,
    float* __restrict__ out)
{
  __shared__ float sH0[9][D];
  __shared__ float sT9[9][H];
  __shared__ float sH [9][D];
  __shared__ float sQ[D];
  __shared__ float sAttn[8];
  __shared__ float sRead[D];

  const int tid = threadIdx.x, wave = tid >> 5, lane = tid & 31;
  const int b = blockIdx.x;

  if (wave < 9) {
    const int pos = (wave < 8) ? idxArr[b * 8 + wave] : (LSEQ - 1);
    const int id  = seq[(long)b * LSEQ + pos];
    sH0[wave][lane]      = embed[id * D + lane];
    sH0[wave][lane + 32] = embed[id * D + lane + 32];
  }
  __syncthreads();
  if (wave < 9) {
#pragma unroll
    for (int uu = 0; uu < 4; ++uu) {
      const int u = lane + uu * 32;
      float t = b1[u];
      for (int k = 0; k < D; ++k) t += sH0[wave][k] * w1[k * H + u];
      sT9[wave][u] = t > 0.0f ? t : 0.0f;
    }
  }
  __syncthreads();
  if (wave < 9) {
    const int n0 = lane, n1 = lane + 32;
    float v0 = b2[n0], v1 = b2[n1];
    for (int k = 0; k < H; ++k) {
      const float tk = sT9[wave][k];
      v0 += tk * w2[k * D + n0];
      v1 += tk * w2[k * D + n1];
    }
    v0 += sH0[wave][n0]; v1 += sH0[wave][n1];
    float s1 = v0 + v1, s2 = v0 * v0 + v1 * v1;
#pragma unroll
    for (int m = 1; m <= 16; m <<= 1) { s1 = wave_xor_add(s1, m); s2 = wave_xor_add(s2, m); }
    const float mu = s1 * (1.0f / D);
    const float var = s2 * (1.0f / D) - mu * mu;
    const float rstd = rsqrtf(var + 1e-5f);
    sH[wave][n0] = (v0 - mu) * rstd * lng[n0] + lnb[n0];
    sH[wave][n1] = (v1 - mu) * rstd * lng[n1] + lnb[n1];
  }
  __syncthreads();

  if (wave == 0) {
    // type_dist = softmax(mean_h @ tp_w + tp_b)
    const int j = lane & 3;
    float tj = tpb[j];
    for (int c = 0; c < D; ++c)
      tj += (gHsum[b * D + c] * (1.0f / LSEQ)) * tpw[c * 4 + j];
    float mx = tj;
    mx = fmaxf(mx, __shfl_xor(mx, 1, 32));
    mx = fmaxf(mx, __shfl_xor(mx, 2, 32));
    const float e = __expf(tj - mx);
    float se = e; se = wave_xor_add(se, 1); se = wave_xor_add(se, 2);
    if (lane < 4) out[NB * VOC + b * 4 + j] = e / se;

    // q = h_last @ q_w + q_b
    float q0 = qb[lane], q1 = qb[lane + 32];
    for (int k = 0; k < D; ++k) {
      const float hk = sH[8][k];
      q0 += hk * qw[k * D + lane];
      q1 += hk * qw[k * D + lane + 32];
    }
    sQ[lane] = q0; sQ[lane + 32] = q1;

    // attention over the 8 memory slots
    const int i = lane & 7;
    float sc = 0.0f;
    for (int c = 0; c < D; ++c) sc += sH[i][c] * sQ[c];
    float mxs = sc;
#pragma unroll
    for (int m = 1; m <= 4; m <<= 1) mxs = fmaxf(mxs, __shfl_xor(mxs, m, 32));
    const float es = __expf(sc - mxs);
    float ss = es;
#pragma unroll
    for (int m = 1; m <= 4; m <<= 1) ss = wave_xor_add(ss, m);
    if (lane < 8) sAttn[i] = es / ss;

    float r0 = 0.0f, r1 = 0.0f;
    for (int i2 = 0; i2 < 8; ++i2) {
      const float a = sAttn[i2];
      r0 += a * sH[i2][lane];
      r1 += a * sH[i2][lane + 32];
    }
    sRead[lane] = r0; sRead[lane + 32] = r1;

    float l0 = ob[lane], l1 = ob[lane + 32];
    for (int c = 0; c < D; ++c) {
      const float rc = sRead[c];
      l0 += rc * ow[c * VOC + lane];
      l1 += rc * ow[c * VOC + lane + 32];
    }
    out[b * VOC + lane]      = l0;
    out[b * VOC + lane + 32] = l1;
  }
}

// ------------------------------- launcher ----------------------------------
extern "C" void kernel_launch(void* const* d_in, const int* in_sizes, int n_in,
                              void* d_out, int out_size, void* d_ws, size_t ws_size,
                              hipStream_t stream) {
  const int*   seq   = (const int*)d_in[0];
  const float* embed = (const float*)d_in[1];
  const float* ff1w  = (const float*)d_in[2];
  const float* ff1b  = (const float*)d_in[3];
  const float* ff2w  = (const float*)d_in[4];
  const float* ff2b  = (const float*)d_in[5];
  const float* lng   = (const float*)d_in[6];
  const float* lnb   = (const float*)d_in[7];
  const float* tpw   = (const float*)d_in[8];
  const float* tpb   = (const float*)d_in[9];
  const float* gatew = (const float*)d_in[12];   // only first 64 rows matter
  const float* qw    = (const float*)d_in[14];
  const float* qb    = (const float*)d_in[15];
  const float* ow    = (const float*)d_in[16];
  const float* ob    = (const float*)d_in[17];

  float* gHsum = (float*)d_ws;                          // B*64 f32
  float* sgate = gHsum + NB * D;                        // B*L f32
  int*   idxA  = (int*)(sgate + (size_t)NB * LSEQ);     // B*8 i32
  float* out   = (float*)d_out;

  k_zero<<<(NB * D + 255) / 256, 256, 0, stream>>>(gHsum, NB * D);
  k_encoder<<<(NB * LSEQ) / TOKB, 128, 0, stream>>>(
      seq, embed, ff1w, ff1b, ff2w, ff2b, lng, lnb, gatew, sgate, gHsum);
  k_topk<<<NB, 256, 0, stream>>>(sgate, idxA);
  k_readout<<<NB, 288, 0, stream>>>(
      seq, embed, ff1w, ff1b, ff2w, ff2b, lng, lnb, tpw, tpb,
      qw, qb, ow, ob, gHsum, idxA, out);
}
